// AttentionRNN_18116172054828
// MI455X (gfx1250) — compile-verified, matrix operand order checked by reading, not run on hardware
//
#include <hip/hip_runtime.h>
#include <cstdint>

#define BATCH   2048
#define SEQ     256
#define HID     32
#define EMB     128
#define GATE    128     // 4*HID
#define NOUT    10
#define DEC_IN  192     // EMB + 2*HID

typedef __bf16 bf16_t;
typedef bf16_t v16bf __attribute__((ext_vector_type(16)));
typedef float  v8f   __attribute__((ext_vector_type(8)));

// ---- branch-free nonlinearities on the native CDNA5 v_tanh_f32 path ----
__device__ __forceinline__ float fast_tanh(float x) {
#if __has_builtin(__builtin_amdgcn_tanhf)
  return __builtin_amdgcn_tanhf(x);               // single V_TANH_F32
#else
  float t = __expf(2.0f * x);                     // branchless fallback: no libm range checks
  return (t - 1.0f) * __builtin_amdgcn_rcpf(t + 1.0f);
#endif
}
__device__ __forceinline__ float sigm(float x) {  // sigmoid via tanh: 1 trans + 1 fma
  return 0.5f * fast_tanh(0.5f * x) + 0.5f;
}

// ---- WMMA fragment builders (CDNA5 16x16x32 bf16 layouts, ISA 7.12.2) ----
// A (16xK tile, row-major in LDS): lane holds row m=lane&15; element e maps to
// k = k0 + 8*(lane>=16) + e + 8*(e>=8)
__device__ __forceinline__ v16bf a_frag_lds(const bf16_t* tile, int stride, int k0, int lane) {
  int m    = lane & 15;
  int koff = (lane >> 4) << 3;
  v16bf a;
  #pragma unroll
  for (int e = 0; e < 16; ++e) {
    int k = k0 + koff + e + ((e >= 8) ? 8 : 0);
    a[e] = tile[m * stride + k];
  }
  return a;
}

// B (32x16 tile) for z = A * W^T:  B[k][n] = W[(jbase+n)*ldw + k0 + k]  (W row-major f32)
__device__ __forceinline__ v16bf b_frag_w(const float* W, int ldw, int jbase, int k0, int lane) {
  int n    = lane & 15;
  int koff = (lane >> 4) << 3;
  v16bf b;
  #pragma unroll
  for (int e = 0; e < 16; ++e) {
    int k = k0 + koff + e + ((e >= 8) ? 8 : 0);
    b[e] = (bf16_t)W[(jbase + n) * ldw + k];
  }
  return b;
}

// ---- Kernel 1: fold embedding through input weights: P[v][j] = emb[v]·W_ih[j] + b ----
__global__ void proj_table_kernel(const float* __restrict__ emb, const float* __restrict__ Wih,
                                  const float* __restrict__ bih, const float* __restrict__ bhh,
                                  float* __restrict__ P) {
  int idx = blockIdx.x * blockDim.x + threadIdx.x;     // 128*128
  int v = idx >> 7, j = idx & 127;
  float acc = bih[j] + bhh[j];
  #pragma unroll
  for (int k = 0; k < HID; ++k) acc += emb[v * HID + k] * Wih[j * HID + k];
  P[idx] = acc;
}

__global__ void zero_kernel(float* __restrict__ p, int n) {
  int i = blockIdx.x * blockDim.x + threadIdx.x;
  if (i < n) p[i] = 0.0f;
}

// ---- Kernel 2: persistent bidirectional LSTM encoder. One wave = 16 batch rows,
// marches through all 256 timesteps. 8 WMMAs/step for h @ W_hh^T. ----
__global__ void __launch_bounds__(32) lstm_encoder_kernel(
    const int* __restrict__ x, const float* __restrict__ Pf, const float* __restrict__ Pb,
    const float* __restrict__ Wf_hh, const float* __restrict__ Wb_hh,
    bf16_t* __restrict__ Abuf) {
  __shared__ int    tok[16 * SEQ];
  __shared__ bf16_t hlds[16 * HID];
  int lane = threadIdx.x;
  int b0   = blockIdx.x * 16;
  int dir  = blockIdx.y;                    // 0 = forward, 1 = backward
  const float* P   = dir ? Pb : Pf;
  const float* Whh = dir ? Wb_hh : Wf_hh;

  for (int i = lane; i < 16 * SEQ; i += 32) {
    int m = i >> 8, s = i & 255;
    tok[i] = x[(b0 + m) * SEQ + s];
  }
  for (int i = lane; i < 16 * HID; i += 32) hlds[i] = (bf16_t)0.0f;
  __syncthreads();

  v16bf bw[8];                              // W_hh^T fragments, resident for whole kernel
  #pragma unroll
  for (int t = 0; t < 8; ++t) bw[t] = b_frag_w(Whh, HID, 16 * t, 0, lane);

  int col = lane & 15, hi = lane >> 4;
  float cst[2][8];
  #pragma unroll
  for (int h2 = 0; h2 < 2; ++h2)
    #pragma unroll
    for (int r = 0; r < 8; ++r) cst[h2][r] = 0.0f;

  v8f zero = {0.f, 0.f, 0.f, 0.f, 0.f, 0.f, 0.f, 0.f};
  #pragma unroll 1
  for (int s = 0; s < SEQ; ++s) {
    int ss = dir ? (SEQ - 1 - s) : s;
    int sn = dir ? (ss - 1) : (ss + 1);     // next timestep (for P-row prefetch)
    v16bf a = a_frag_lds(hlds, HID, 0, lane);
    v8f z[8];
    #pragma unroll
    for (int t = 0; t < 8; ++t)
      z[t] = __builtin_amdgcn_wmma_f32_16x16x32_bf16(false, a, false, bw[t],
                                                     (short)0, zero, false, false);
    __syncthreads();                        // done reading hlds for this step
    #pragma unroll
    for (int r = 0; r < 8; ++r) {
      int m = r + 8 * hi;
      const float* prow = P + tok[m * SEQ + ss] * GATE;
      if (s + 1 < SEQ) {                    // hide next step's gather under this step's math
        const float* pnext = P + tok[m * SEQ + sn] * GATE;
        __builtin_prefetch(pnext + ((lane & 3) * 32), 0, 3);
      }
      #pragma unroll
      for (int h2 = 0; h2 < 2; ++h2) {
        int nn = h2 * 16 + col;
        float iv = z[0 + h2][r] + prow[0  + nn];
        float fv = z[2 + h2][r] + prow[32 + nn];
        float gv = z[4 + h2][r] + prow[64 + nn];
        float ov = z[6 + h2][r] + prow[96 + nn];
        float c  = sigm(fv) * cst[h2][r] + sigm(iv) * fast_tanh(gv);
        float h  = sigm(ov) * fast_tanh(c);
        cst[h2][r] = c;
        hlds[m * HID + nn] = (bf16_t)h;     // transpose D-layout -> A-layout via LDS
        Abuf[(((size_t)(b0 + m)) * SEQ + ss) * (2 * HID) + dir * HID + nn] = (bf16_t)h;
      }
    }
    __syncthreads();
  }
}

// ---- Kernel 3: a_proj[b][s] = a[b][s]·w_a + b_att ----
__global__ void aproj_kernel(const bf16_t* __restrict__ Abuf, const float* __restrict__ w_att,
                             const float* __restrict__ b_att, float* __restrict__ aproj) {
  int idx = blockIdx.x * blockDim.x + threadIdx.x;     // B*S
  const bf16_t* row = Abuf + (size_t)idx * (2 * HID);
  float acc = b_att[0];
  #pragma unroll
  for (int e = 0; e < 2 * HID; ++e) acc += (float)row[e] * w_att[HID + e];
  aproj[idx] = acc;
}

// ---- Kernel 4: per-step attention: softmax over S, ctx = alpha·a. One block per b. ----
__global__ void __launch_bounds__(256) attn_kernel(
    const bf16_t* __restrict__ Abuf, const float* __restrict__ aproj,
    const float* __restrict__ hdec, const float* __restrict__ w_att,
    float* __restrict__ ctx) {
  int b = blockIdx.x, tid = threadIdx.x;
  __shared__ float red[256];
  __shared__ float alp[256];
  float hd = 0.f;
  #pragma unroll
  for (int k = 0; k < HID; ++k) hd += hdec[b * HID + k] * w_att[k];
  float sc = aproj[b * SEQ + tid] + hd;
  red[tid] = sc; __syncthreads();
  for (int off = 128; off > 0; off >>= 1) {
    if (tid < off) red[tid] = fmaxf(red[tid], red[tid + off]);
    __syncthreads();
  }
  float mx = red[0]; __syncthreads();
  float ex = __expf(sc - mx);
  alp[tid] = ex; red[tid] = ex; __syncthreads();
  for (int off = 128; off > 0; off >>= 1) {
    if (tid < off) red[tid] += red[tid + off];
    __syncthreads();
  }
  float inv = __builtin_amdgcn_rcpf(red[0]);   // v_rcp_f32, no IEEE div sequence
  __syncthreads();
  if (tid < 2 * HID) {                      // thread e accumulates ctx[b][e]; coalesced over e
    float acc = 0.f;
    const bf16_t* base = Abuf + (size_t)b * SEQ * (2 * HID) + tid;
    #pragma unroll 4
    for (int s = 0; s < SEQ; ++s) acc += alp[s] * (float)base[(size_t)s * (2 * HID)];
    ctx[b * (2 * HID) + tid] = acc * inv;
  }
}

// ---- Kernel 5: decoder LSTM cell + output projection, WMMA per 16-batch tile ----
__global__ void __launch_bounds__(32) dec_cell_kernel(
    float* __restrict__ out, const float* __restrict__ ctx,
    float* __restrict__ hdec, float* __restrict__ cdec,
    const float* __restrict__ Wd_ih, const float* __restrict__ Wd_hh,
    const float* __restrict__ bd_ih, const float* __restrict__ bd_hh,
    const float* __restrict__ W_out, const float* __restrict__ b_out, int step) {
  __shared__ bf16_t inp[16 * DEC_IN];
  __shared__ bf16_t hl[16 * HID];
  int lane = threadIdx.x;
  int b0   = blockIdx.x * 16;

  for (int i = lane; i < 16 * DEC_IN; i += 32) {       // inp = [py (128) | ctx (64)]
    int m = i / DEC_IN, e = i % DEC_IN;
    float v;
    if (e < EMB) v = (step == 0) ? 0.0f : out[((size_t)(b0 + m) * NOUT + (step - 1)) * EMB + e];
    else         v = ctx[(b0 + m) * (2 * HID) + (e - EMB)];
    inp[i] = (bf16_t)v;
  }
  for (int i = lane; i < 16 * HID; i += 32)
    hl[i] = (bf16_t)hdec[(b0 + (i >> 5)) * HID + (i & 31)];
  __syncthreads();

  int col = lane & 15, hi = lane >> 4;
  v8f zero = {0.f, 0.f, 0.f, 0.f, 0.f, 0.f, 0.f, 0.f};
  v8f z[8];
  #pragma unroll
  for (int t = 0; t < 8; ++t) {
    v8f acc = zero;
    #pragma unroll
    for (int kc = 0; kc < 6; ++kc) {                   // K = 192 in 6 chunks of 32
      v16bf a = a_frag_lds(inp, DEC_IN, kc * 32, lane);
      v16bf w = b_frag_w(Wd_ih, DEC_IN, 16 * t, kc * 32, lane);
      acc = __builtin_amdgcn_wmma_f32_16x16x32_bf16(false, a, false, w, (short)0, acc, false, false);
    }
    v16bf ah = a_frag_lds(hl, HID, 0, lane);
    v16bf wh = b_frag_w(Wd_hh, HID, 16 * t, 0, lane);
    acc = __builtin_amdgcn_wmma_f32_16x16x32_bf16(false, ah, false, wh, (short)0, acc, false, false);
    int j = 16 * t + col;
    float bd = bd_ih[j] + bd_hh[j];
    #pragma unroll
    for (int r = 0; r < 8; ++r) acc[r] += bd;
    z[t] = acc;
  }
  __syncthreads();
  #pragma unroll
  for (int r = 0; r < 8; ++r) {
    int m = r + 8 * hi;
    #pragma unroll
    for (int h2 = 0; h2 < 2; ++h2) {
      int nn = h2 * 16 + col;
      float iv = z[0 + h2][r], fv = z[2 + h2][r], gv = z[4 + h2][r], ov = z[6 + h2][r];
      float cold = cdec[(b0 + m) * HID + nn];
      float c = sigm(fv) * cold + sigm(iv) * fast_tanh(gv);
      float h = sigm(ov) * fast_tanh(c);
      cdec[(b0 + m) * HID + nn] = c;
      hdec[(b0 + m) * HID + nn] = h;
      hl[m * HID + nn] = (bf16_t)h;
    }
  }
  __syncthreads();
  #pragma unroll
  for (int t = 0; t < 8; ++t) {                        // py2 = h2 @ W_out^T + b_out
    v16bf a = a_frag_lds(hl, HID, 0, lane);
    v16bf w = b_frag_w(W_out, HID, 16 * t, 0, lane);
    v8f p = __builtin_amdgcn_wmma_f32_16x16x32_bf16(false, a, false, w, (short)0, zero, false, false);
    int j = 16 * t + col;
    float bo = b_out[j];
    #pragma unroll
    for (int r = 0; r < 8; ++r) {
      int m = r + 8 * hi;
      out[((size_t)(b0 + m) * NOUT + step) * EMB + j] = p[r] + bo;
    }
  }
}

extern "C" void kernel_launch(void* const* d_in, const int* in_sizes, int n_in,
                              void* d_out, int out_size, void* d_ws, size_t ws_size,
                              hipStream_t stream) {
  const int*   x     = (const int*)  d_in[0];
  const float* emb   = (const float*)d_in[2];
  const float* Wf_ih = (const float*)d_in[3];
  const float* Wf_hh = (const float*)d_in[4];
  const float* bf_ih = (const float*)d_in[5];
  const float* bf_hh = (const float*)d_in[6];
  const float* Wb_ih = (const float*)d_in[7];
  const float* Wb_hh = (const float*)d_in[8];
  const float* bb_ih = (const float*)d_in[9];
  const float* bb_hh = (const float*)d_in[10];
  const float* Wd_ih = (const float*)d_in[11];
  const float* Wd_hh = (const float*)d_in[12];
  const float* bd_ih = (const float*)d_in[13];
  const float* bd_hh = (const float*)d_in[14];
  const float* w_att = (const float*)d_in[15];
  const float* b_att = (const float*)d_in[16];
  const float* W_out = (const float*)d_in[17];
  const float* b_out = (const float*)d_in[18];
  float* out = (float*)d_out;

  char* ws = (char*)d_ws;
  size_t off = 0;
  auto alloc = [&](size_t bytes) -> void* {
    void* p = ws + off;
    off += (bytes + 255) & ~(size_t)255;
    return p;
  };
  float*  Pf    = (float*) alloc(EMB * GATE * sizeof(float));            // 64 KB
  float*  Pb    = (float*) alloc(EMB * GATE * sizeof(float));            // 64 KB
  bf16_t* Abuf  = (bf16_t*)alloc((size_t)BATCH * SEQ * 2 * HID * 2);     // 67 MB bf16
  float*  aproj = (float*) alloc((size_t)BATCH * SEQ * sizeof(float));   // 2 MB
  float*  hdec  = (float*) alloc((size_t)BATCH * HID * sizeof(float));   // 256 KB
  float*  cdec  = (float*) alloc((size_t)BATCH * HID * sizeof(float));   // 256 KB (adjacent)
  float*  ctxb  = (float*) alloc((size_t)BATCH * 2 * HID * sizeof(float));

  // 1. fold emb through input weights (kills the per-step input GEMM)
  proj_table_kernel<<<(EMB * GATE) / 256, 256, 0, stream>>>(emb, Wf_ih, bf_ih, bf_hh, Pf);
  proj_table_kernel<<<(EMB * GATE) / 256, 256, 0, stream>>>(emb, Wb_ih, bb_ih, bb_hh, Pb);
  // 2. zero decoder state (hdec & cdec are contiguous)
  zero_kernel<<<(2 * BATCH * HID + 255) / 256, 256, 0, stream>>>(hdec, 2 * BATCH * HID);
  // 3. persistent bidirectional encoder: 128 tiles x 2 dirs, one wave each
  lstm_encoder_kernel<<<dim3(BATCH / 16, 2), 32, 0, stream>>>(x, Pf, Pb, Wf_hh, Wb_hh, Abuf);
  // 4. attention pre-projection
  aproj_kernel<<<(BATCH * SEQ) / 256, 256, 0, stream>>>(Abuf, w_att, b_att, aproj);
  // 5. 10 decoder steps (n_output fixed at 10 in the reference setup)
  for (int t = 0; t < NOUT; ++t) {
    attn_kernel<<<BATCH, 256, 0, stream>>>(Abuf, aproj, hdec, w_att, ctxb);
    dec_cell_kernel<<<BATCH / 16, 32, 0, stream>>>(out, ctxb, hdec, cdec,
                                                   Wd_ih, Wd_hh, bd_ih, bd_hh,
                                                   W_out, b_out, t);
  }
}
